// TabNet_77867757077109
// MI455X (gfx1250) — matrix operand order repaired and checked by reading, not compiled, not run on hardware
//
#include <hip/hip_runtime.h>
#include <hip/hip_bf16.h>
#include <math.h>

// ---------------------------------------------------------------------------
// TabNet forward for MI455X (gfx1250, wave32, WMMA).
// All GEMMs use v_wmma_f32_16x16x32_bf16 (f32 accumulate), double-buffered
// LDS tiles, vectorized f32->bf16 staging.
// ---------------------------------------------------------------------------

typedef __attribute__((ext_vector_type(16))) __bf16 v16bf;
typedef __attribute__((ext_vector_type(8)))  __bf16 v8bf;
typedef __attribute__((ext_vector_type(2)))  __bf16 v2bf;
typedef __attribute__((ext_vector_type(8)))  float  v8f;
typedef __attribute__((ext_vector_type(4)))  float  v4f;

union BF16x16 { v16bf v; v8bf h[2]; };

#define BN_EPS_F     1e-5f
#define SQRT_HALF_F  0.70710678118654752440f
#define RELAX_F      2.0f

// ---------------------------------------------------------------------------
// GEMM: C[M x NC] = A[M x K] * W[K x NC], f32 in memory, bf16 WMMA math.
// Block tile: 128 rows x 64 cols, 256 threads = 8 wave32s; wave w owns the
// 16-row strip w as 4 accumulators of 16x16. K-step 32, double-buffered LDS.
// A tile row-major (k contiguous); B tile transposed [n][k]; rows padded to
// 40 bf16 (80B) to break bank-conflict strides while keeping 16B alignment.
// Fragment layouts follow CDNA5 ISA 7.12.2 (wave32).
// ---------------------------------------------------------------------------
#define GEMM_TM 128
#define GEMM_TN 64
#define GEMM_KS 32
#define GEMM_KA 40   // padded row length

__device__ __forceinline__
void stage_tile(__bf16 (*As)[GEMM_KA], __bf16 (*Bs)[GEMM_KA],
                const float* __restrict__ A, int lda,
                const float* __restrict__ W, int ldw,
                int block_m, int block_n, int k0, int tid)
{
    // ---- A tile: 128x32, 16 contiguous floats per thread ----
    {
        const int r  = tid >> 1;
        const int c0 = (tid & 1) << 4;
        const float* ap = A + (long)(block_m + r) * lda + (k0 + c0);
        v4f a0 = *(const v4f*)(ap + 0);
        v4f a1 = *(const v4f*)(ap + 4);
        v4f a2 = *(const v4f*)(ap + 8);
        v4f a3 = *(const v4f*)(ap + 12);
        v8bf p0, p1;
        #pragma unroll
        for (int j = 0; j < 4; ++j) {
            p0[j]     = (__bf16)a0[j];
            p0[4 + j] = (__bf16)a1[j];
            p1[j]     = (__bf16)a2[j];
            p1[4 + j] = (__bf16)a3[j];
        }
        *(v8bf*)&As[r][c0]     = p0;   // 16B-aligned ds_store_b128
        *(v8bf*)&As[r][c0 + 8] = p1;
    }
    // ---- B tile: 32x64 -> transposed [n][k]; thread covers k-pair x 4 n ----
    {
        const int kp = tid >> 4;          // 0..15 : k = 2kp, 2kp+1
        const int n0 = (tid & 15) << 2;   // 0..60
        const float* w0 = W + (long)(k0 + 2 * kp) * ldw + (block_n + n0);
        const float* w1 = w0 + ldw;
        v4f b0 = *(const v4f*)w0;         // coalesced along n
        v4f b1 = *(const v4f*)w1;
        #pragma unroll
        for (int j = 0; j < 4; ++j) {
            v2bf p;
            p[0] = (__bf16)b0[j];         // k = 2kp
            p[1] = (__bf16)b1[j];         // k = 2kp+1
            *(v2bf*)&Bs[n0 + j][2 * kp] = p;   // packed ds_store_b32
        }
    }
}

__global__ __launch_bounds__(256)
void gemm_bf16_wmma(const float* __restrict__ A, int lda,
                    const float* __restrict__ W, int ldw,
                    float* __restrict__ C, int ldc,
                    int M, int K, int NC)
{
    __shared__ __bf16 As[2][GEMM_TM][GEMM_KA];   // 20 KB
    __shared__ __bf16 Bs[2][GEMM_TN][GEMM_KA];   // 10 KB

    const int block_m = blockIdx.x * GEMM_TM;
    const int block_n = blockIdx.y * GEMM_TN;
    const int tid  = threadIdx.x;
    const int wave = tid >> 5;        // 0..7
    const int lane = tid & 31;
    const int hi   = lane >> 4;       // lane half
    const int lo16 = lane & 15;

    v8f acc[4] = {};
    const int nk = K / GEMM_KS;

    stage_tile(As[0], Bs[0], A, lda, W, ldw, block_m, block_n, 0, tid);
    __syncthreads();

    for (int kt = 0; kt < nk; ++kt) {
        const int cur = kt & 1;
        if (kt + 1 < nk) {
            stage_tile(As[cur ^ 1], Bs[cur ^ 1], A, lda, W, ldw,
                       block_m, block_n, (kt + 1) * GEMM_KS, tid);
        }
        if (kt + 2 < nk) {
            __builtin_prefetch(A + (long)(block_m + (tid >> 1)) * lda + (kt + 2) * GEMM_KS, 0, 1);
        }

        // A fragment (16-bit A 16x32): j<8 -> K=hi*8+j ; j>=8 -> K=16+hi*8+(j-8)
        BF16x16 afrag;
        afrag.h[0] = *(const v8bf*)&As[cur][wave * 16 + lo16][hi * 8];
        afrag.h[1] = *(const v8bf*)&As[cur][wave * 16 + lo16][16 + hi * 8];

        #pragma unroll
        for (int t = 0; t < 4; ++t) {
            // B fragment (16-bit B 32x16): n = lane%16, j<8 -> K=hi*8+j ; j>=8 -> K=16+hi*8+(j-8)
            BF16x16 bfrag;
            const int ncol = t * 16 + lo16;
            bfrag.h[0] = *(const v8bf*)&Bs[cur][ncol][hi * 8];
            bfrag.h[1] = *(const v8bf*)&Bs[cur][ncol][16 + hi * 8];
            acc[t] = __builtin_amdgcn_wmma_f32_16x16x32_bf16(
                false, afrag.v, false, bfrag.v, (short)0, acc[t], false, false);
        }
        __syncthreads();
    }

    // write out. C/D layout: lane -> n = lane%16 ; vgpr v -> m = v + 8*(lane/16)
    #pragma unroll
    for (int t = 0; t < 4; ++t) {
        #pragma unroll
        for (int v = 0; v < 8; ++v) {
            int gr = block_m + wave * 16 + hi * 8 + v;
            int gc = block_n + t * 16 + lo16;
            C[(long)gr * ldc + gc] = acc[t][v];
        }
    }
}

// ---------------------------------------------------------------------------
// Column stats: sums[c] += sum over rows, sums[NC+c] += sum of squares.
// ---------------------------------------------------------------------------
__global__ void colstats_kernel(const float* __restrict__ X, int ldx,
                                int M, int NC, float* __restrict__ sums)
{
    const int rpb = (M + gridDim.x - 1) / gridDim.x;
    const int r0 = blockIdx.x * rpb;
    int r1 = r0 + rpb; if (r1 > M) r1 = M;
    for (int c = threadIdx.x; c < NC; c += blockDim.x) {
        float s = 0.f, s2 = 0.f;
        for (int r = r0; r < r1; ++r) {
            float v = X[(long)r * ldx + c];
            s += v; s2 += v * v;
        }
        atomicAdd(&sums[c], s);
        atomicAdd(&sums[NC + c], s2);
    }
}

__global__ void zero_f32(float* __restrict__ p, long n)
{
    long i = (long)blockIdx.x * blockDim.x + threadIdx.x;
    if (i < n) p[i] = 0.f;
}

__global__ void init_mp(const float* __restrict__ feat,
                        float* __restrict__ masked, float* __restrict__ prior, long n)
{
    long i = (long)blockIdx.x * blockDim.x + threadIdx.x;
    if (i < n) { masked[i] = feat[i]; prior[i] = 1.f; }
}

// ---------------------------------------------------------------------------
// h = GLU(BN(xh))                      (residual==0)
// h = (h + GLU(BN(xh))) * sqrt(0.5)    (residual==1)
// xh is N x 256, h is N x 128. stats = [sum(256), sumsq(256)].
// ---------------------------------------------------------------------------
__global__ void glu_bn_kernel(const float* __restrict__ xh,
                              const float* __restrict__ stats,
                              const float* __restrict__ g, const float* __restrict__ b,
                              float* __restrict__ h, int M, int residual)
{
    long i = (long)blockIdx.x * blockDim.x + threadIdx.x;
    if (i >= (long)M * 128) return;
    const int row = (int)(i >> 7);
    const int col = (int)(i & 127);
    const float invN = 1.f / (float)M;

    float mu_a = stats[col] * invN;
    float va   = stats[256 + col] * invN - mu_a * mu_a;
    float xa = (xh[(long)row * 256 + col] - mu_a) * rsqrtf(va + BN_EPS_F) * g[col] + b[col];

    const int cb = col + 128;
    float mu_b = stats[cb] * invN;
    float vb   = stats[256 + cb] * invN - mu_b * mu_b;
    float xb = (xh[(long)row * 256 + cb] - mu_b) * rsqrtf(vb + BN_EPS_F) * g[cb] + b[cb];

    float gl = xa * (1.f / (1.f + expf(-xb)));
    h[i] = residual ? (h[i] + gl) * SQRT_HALF_F : gl;
}

// agg[row][d] += relu(h[row][d]),  d < 64 (dec = first 64 cols of h)
__global__ void agg_relu_kernel(const float* __restrict__ h, float* __restrict__ agg, int M)
{
    long i = (long)blockIdx.x * blockDim.x + threadIdx.x;
    if (i >= (long)M * 64) return;
    int row = (int)(i >> 6), d = (int)(i & 63);
    float v = h[(long)row * 128 + d];
    agg[i] += fmaxf(v, 0.f);
}

// ---------------------------------------------------------------------------
// Attention: one wave32 per row of 512 (16 elems/lane).
//   z = BN(raw)*prior; z -= rowmax; solve sum(relu(z-tau))=1 by bisection
//   (exact sparsemax condition); m = relu(z-tau); prior *= (RELAX - m);
//   masked = m * feature; sparsity += mean entropy.
// ---------------------------------------------------------------------------
__global__ __launch_bounds__(256)
void attn_sparsemax_kernel(float* __restrict__ m_io,
                           const float* __restrict__ stats,   // [sum(512), sumsq(512)]
                           const float* __restrict__ g, const float* __restrict__ b,
                           const float* __restrict__ feature,
                           float* __restrict__ prior,
                           float* __restrict__ masked,
                           float* __restrict__ sparsity, int M)
{
    const int wave = threadIdx.x >> 5;
    const int lane = threadIdx.x & 31;
    const int row = blockIdx.x * 8 + wave;
    if (row >= M) return;
    const float invN = 1.f / (float)M;
    const long base = (long)row * 512;

    float z[16];
    #pragma unroll
    for (int j = 0; j < 16; ++j) {
        int c = lane + 32 * j;                    // coalesced
        float mu = stats[c] * invN;
        float va = stats[512 + c] * invN - mu * mu;
        float x = (m_io[base + c] - mu) * rsqrtf(va + BN_EPS_F) * g[c] + b[c];
        z[j] = x * prior[base + c];
    }
    // row max (wave32 reduction)
    float mx = z[0];
    #pragma unroll
    for (int j = 1; j < 16; ++j) mx = fmaxf(mx, z[j]);
    #pragma unroll
    for (int off = 16; off >= 1; off >>= 1) mx = fmaxf(mx, __shfl_xor(mx, off, 32));
    #pragma unroll
    for (int j = 0; j < 16; ++j) z[j] -= mx;

    // bisection: f(tau) = sum relu(z - tau), tau* in [-1, 0], f(tau*) = 1
    float lo = -1.f, hi = 0.f;
    for (int it = 0; it < 30; ++it) {
        float tau = 0.5f * (lo + hi);
        float s = 0.f;
        #pragma unroll
        for (int j = 0; j < 16; ++j) s += fmaxf(z[j] - tau, 0.f);
        #pragma unroll
        for (int off = 16; off >= 1; off >>= 1) s += __shfl_xor(s, off, 32);
        if (s > 1.f) lo = tau; else hi = tau;
    }
    const float tau = 0.5f * (lo + hi);

    float rowsum = 0.f;
    #pragma unroll
    for (int j = 0; j < 16; ++j) {
        int c = lane + 32 * j;
        float mv = fmaxf(z[j] - tau, 0.f);
        m_io[base + c] = mv;
        prior[base + c] = prior[base + c] * (RELAX_F - mv);
        masked[base + c] = mv * feature[base + c];
        rowsum += mv * logf(mv + 1e-5f);
    }
    #pragma unroll
    for (int off = 16; off >= 1; off >>= 1) rowsum += __shfl_xor(rowsum, off, 32);
    if (lane == 0) atomicAdd(sparsity, rowsum * invN * 0.25f);   // /N /(STEPS-1)
}

// ---------------------------------------------------------------------------
// Host orchestration
// ---------------------------------------------------------------------------
static inline void run_gemm(const float* A, int lda, const float* W, int ldw,
                            float* C, int ldc, int M, int K, int NC, hipStream_t s)
{
    dim3 grid(M / GEMM_TM, NC / GEMM_TN);
    gemm_bf16_wmma<<<grid, 256, 0, s>>>(A, lda, W, ldw, C, ldc, M, K, NC);
}

static inline void run_stats(const float* X, int ldx, int M, int NC, float* stats, hipStream_t s)
{
    zero_f32<<<(2 * NC + 255) / 256, 256, 0, s>>>(stats, 2 * NC);
    colstats_kernel<<<256, 256, 0, s>>>(X, ldx, M, NC, stats);
}

extern "C" void kernel_launch(void* const* d_in, const int* in_sizes, int n_in,
                              void* d_out, int out_size, void* d_ws, size_t ws_size,
                              hipStream_t stream)
{
    constexpr int F_ = 512, H2_ = 256, OUT_ = 128, STEPS_ = 5;

    const float* feat     = (const float*)d_in[0];
    const float* sh_W1    = (const float*)d_in[1];
    const float* sh_bn1_g = (const float*)d_in[2];
    const float* sh_bn1_b = (const float*)d_in[3];
    const float* sh_W2    = (const float*)d_in[4];
    const float* sh_bn2_g = (const float*)d_in[5];
    const float* sh_bn2_b = (const float*)d_in[6];
    const float* ft_W     = (const float*)d_in[7];   // (5,2,128,256)
    const float* ft_g     = (const float*)d_in[8];   // (5,2,256)
    const float* ft_b     = (const float*)d_in[9];
    const float* at_W     = (const float*)d_in[10];  // (4,64,512)
    const float* at_g     = (const float*)d_in[11];  // (4,512)
    const float* at_b     = (const float*)d_in[12];
    const float* fc_W     = (const float*)d_in[13];  // (64,128)

    const int N = in_sizes[0] / F_;                  // 65536

    // workspace layout
    float* masked = (float*)d_ws;                    // N*512
    float* prior  = masked + (size_t)N * F_;         // N*512
    float* xh     = prior  + (size_t)N * F_;         // N*256
    float* h      = xh     + (size_t)N * H2_;        // N*128
    float* agg    = h      + (size_t)N * 128;        // N*64
    float* stats  = agg    + (size_t)N * 64;         // 1024

    // output layout: logits (N*128) | masks (4*N*512) | sparsity (1)
    float* logits   = (float*)d_out;
    float* masks    = logits + (size_t)N * OUT_;
    float* sparsity = masks + (size_t)(STEPS_ - 1) * N * F_;

    const long NF = (long)N * F_;
    init_mp<<<(unsigned)((NF + 255) / 256), 256, 0, stream>>>(feat, masked, prior, NF);
    zero_f32<<<(unsigned)(((long)N * 64 + 255) / 256), 256, 0, stream>>>(agg, (long)N * 64);
    zero_f32<<<1, 256, 0, stream>>>(sparsity, 1);

    const unsigned gElem = (unsigned)(((long)N * 128 + 255) / 256);

    for (int step = 0; step < STEPS_; ++step) {
        // shared block 1: xh = masked @ sh_W1 ; h = GLU(BN(xh))
        run_gemm(masked, F_, sh_W1, H2_, xh, H2_, N, F_, H2_, stream);
        run_stats(xh, H2_, N, H2_, stats, stream);
        glu_bn_kernel<<<gElem, 256, 0, stream>>>(xh, stats, sh_bn1_g, sh_bn1_b, h, N, 0);

        // shared block 2: residual GLU
        run_gemm(h, 128, sh_W2, H2_, xh, H2_, N, 128, H2_, stream);
        run_stats(xh, H2_, N, H2_, stats, stream);
        glu_bn_kernel<<<gElem, 256, 0, stream>>>(xh, stats, sh_bn2_g, sh_bn2_b, h, N, 1);

        // step-dependent feature transformer blocks
        for (int j = 0; j < 2; ++j) {
            const float* W = ft_W + (size_t)(step * 2 + j) * 128 * H2_;
            const float* g = ft_g + (size_t)(step * 2 + j) * H2_;
            const float* b = ft_b + (size_t)(step * 2 + j) * H2_;
            run_gemm(h, 128, W, H2_, xh, H2_, N, 128, H2_, stream);
            run_stats(xh, H2_, N, H2_, stats, stream);
            glu_bn_kernel<<<gElem, 256, 0, stream>>>(xh, stats, g, b, h, N, 1);
        }

        if (step != 0) {
            agg_relu_kernel<<<(unsigned)(((long)N * 64 + 255) / 256), 256, 0, stream>>>(h, agg, N);
        }

        if (step != STEPS_ - 1) {
            float* mslot = masks + (size_t)step * N * F_;
            // coef = h[:, 64:128] ; mslot = coef @ at_W[step]
            run_gemm(h + 64, 128, at_W + (size_t)step * 64 * F_, F_, mslot, F_, N, 64, F_, stream);
            run_stats(mslot, F_, N, F_, stats, stream);
            attn_sparsemax_kernel<<<(N + 7) / 8, 256, 0, stream>>>(
                mslot, stats, at_g + (size_t)step * F_, at_b + (size_t)step * F_,
                feat, prior, masked, sparsity, N);
        }
    }

    // logits = agg @ fc_W
    run_gemm(agg, 64, fc_W, OUT_, logits, OUT_, N, 64, OUT_, stream);
}
